// CPEN_15625091023092
// MI455X (gfx1250) — matrix operand order repaired and checked by reading, not compiled, not run on hardware
//
#include <hip/hip_runtime.h>
#include <hip/hip_bf16.h>

typedef __attribute__((ext_vector_type(16))) _Float16 v16h;
typedef __attribute__((ext_vector_type(8)))  float    v8f;
typedef __attribute__((ext_vector_type(2)))  float    v2f;

#define EPS_REG 1e-3f

constexpr int ilog2c(int v) { return v == 1 ? 0 : 1 + ilog2c(v / 2); }

// ---------------------------------------------------------------------------
// Weight pack: OIHW f32 [32][32][3][3] -> f16 A fragments, contiguous k per lane half
// packed[((tile*9+tap)*32 + lane)*16 + j], A 16x32 f16, contiguous-k order used here (same as B loads):
//   m = tile*16 + (lane&15)
//   half = lane>>4 ; v=j>>1 ; pos=j&1
//   k = half*16 + j
// ---------------------------------------------------------------------------
__global__ void pack_w32_kernel(const float* __restrict__ w, _Float16* __restrict__ dst) {
    int i = blockIdx.x * 256 + threadIdx.x;
    if (i >= 2 * 9 * 32 * 16) return;
    int j    = i & 15;
    int lane = (i >> 4) & 31;
    int tt   = i >> 9;            // tile*9 + tap
    int tap  = tt % 9;
    int tile = tt / 9;
    int half = lane >> 4;
    int v    = j >> 1;
    int pos  = j & 1;
    int k    = half * 16 + j;
    int m    = tile * 16 + (lane & 15);
    dst[i] = (_Float16)w[(m * 32 + k) * 9 + tap];
}

// ---------------------------------------------------------------------------
// Stem conv: 1->32, 3x3, SAME, ReLU.
// raw f32 [4][256][256] -> f16 HWC with 1-px zero halo: [4][258][258][32]
// ---------------------------------------------------------------------------
__global__ void stem_kernel(const float* __restrict__ raw,
                            const float* __restrict__ w,   // [32][1][3][3]
                            const float* __restrict__ b,
                            _Float16* __restrict__ out) {
    __shared__ float lw[288];
    __shared__ float lb[32];
    int tid = threadIdx.x;
    if (tid < 288) lw[tid] = w[tid];
    if (tid < 32)  lb[tid] = b[tid];
    __syncthreads();
    int img = blockIdx.y;
    int pix = blockIdx.x * 256 + tid;
    int x = pix & 255, y = pix >> 8;
    const float* rimg = raw + (size_t)img * 65536;
    float acc[32];
#pragma unroll
    for (int c = 0; c < 32; ++c) acc[c] = lb[c];
#pragma unroll
    for (int t = 0; t < 9; ++t) {
        int dy = t / 3 - 1, dx = t % 3 - 1;
        int xx = x + dx, yy = y + dy;
        float v = (xx >= 0 && xx < 256 && yy >= 0 && yy < 256) ? rimg[yy * 256 + xx] : 0.0f;
#pragma unroll
        for (int c = 0; c < 32; ++c) acc[c] += v * lw[c * 9 + t];
    }
    alignas(16) _Float16 h[32];
#pragma unroll
    for (int c = 0; c < 32; ++c) h[c] = (_Float16)fmaxf(acc[c], 0.0f);
    uint4* dst = (uint4*)(out + ((size_t)img * 258 * 258 + (size_t)(y + 1) * 258 + (x + 1)) * 32);
    const uint4* src = (const uint4*)h;
#pragma unroll
    for (int q = 0; q < 4; ++q) dst[q] = src[q];
}

// ---------------------------------------------------------------------------
// 32->32 3x3 conv as implicit GEMM on v_wmma_f32_16x16x32_f16.
// Templated geometry -> tap offsets become load immediates; zero-halo input
// -> no bounds checks, fully straight-line inner loop (36 wmma / wave).
// ---------------------------------------------------------------------------
template <int WI, int STRIDE, bool OUTF32, bool RELU>
__global__ __launch_bounds__(128)
void conv32_wmma_kernel(const _Float16* __restrict__ inAct,
                        const _Float16* __restrict__ packW,
                        const float* __restrict__ bias,
                        _Float16* __restrict__ outH,
                        float* __restrict__ outF) {
    constexpr int WO     = (STRIDE == 2) ? WI / 2 : WI;
    constexpr int WPI    = WI + 2;
    constexpr int WPO    = WO + 2;
    constexpr int OFF0   = (STRIDE == 2) ? 1 : 0;   // padded-coord origin offset
    constexpr int WOSH   = ilog2c(WO);

    int img  = blockIdx.y;
    int lane = threadIdx.x & 31;
    int wave = threadIdx.x >> 5;
    int coutTile = wave >> 1;
    int ph   = wave & 1;
    int tileBase = blockIdx.x * 128;
    const _Float16* inImg = inAct + (size_t)img * WPI * WPI * 32;
    int kg = lane >> 4, nsel = lane & 15;

    // A fragments (weights) for this Cout tile: 9 taps x v16h, in registers.
    v16h a[9];
    const v16h* pw = (const v16h*)packW;
#pragma unroll
    for (int t = 0; t < 9; ++t) a[t] = pw[(coutTile * 9 + t) * 32 + lane];

    // Per-subtile base pointers (hoisted out of the tap loop).
    const _Float16* bp[4];
    int ox_[4], oy_[4], opix_[4];
#pragma unroll
    for (int s4 = 0; s4 < 4; ++s4) {
        int opix = tileBase + (ph * 4 + s4) * 16 + nsel;
        int ox = opix & (WO - 1);
        int oy = opix >> WOSH;
        opix_[s4] = opix; ox_[s4] = ox; oy_[s4] = oy;
        bp[s4] = inImg + ((size_t)(oy * STRIDE + OFF0) * WPI + (size_t)(ox * STRIDE + OFF0)) * 32
                 + kg * 16;
    }
    // Pull the next rows toward the caches (global_prefetch_b8).
    __builtin_prefetch((const void*)(bp[3] + (size_t)3 * WPI * 32), 0, 0);

    v8f zero8 = {};
    v8f acc[4] = {zero8, zero8, zero8, zero8};

#pragma unroll
    for (int t = 0; t < 9; ++t) {
        const int dy = t / 3, dx = t % 3;
#pragma unroll
        for (int s4 = 0; s4 < 4; ++s4) {
            v16h bf = *(const v16h*)(bp[s4] + (dy * WPI + dx) * 32);   // immediate offset
            acc[s4] = __builtin_amdgcn_wmma_f32_16x16x32_f16(
                false, a[t], false, bf, (short)0, acc[s4], false, false);
        }
    }

    // Epilogue: D(m,n) at lane = n + 16*(m>>3), vgpr = m&7  (m within 16-row tile)
    int cbase = coutTile * 16 + kg * 8;
    float bv[8];
#pragma unroll
    for (int v = 0; v < 8; ++v) bv[v] = bias[cbase + v];

#pragma unroll
    for (int s4 = 0; s4 < 4; ++s4) {
        if (OUTF32) {
            size_t ibase = (size_t)img * 32 * (size_t)(WO * WO);
#pragma unroll
            for (int v = 0; v < 8; ++v) {
                float x = acc[s4][v] + bv[v];
                if (RELU) x = fmaxf(x, 0.0f);
                outF[ibase + (size_t)(cbase + v) * (WO * WO) + opix_[s4]] = x;
            }
        } else {
            alignas(16) _Float16 h8[8];
#pragma unroll
            for (int v = 0; v < 8; ++v) {
                float x = acc[s4][v] + bv[v];
                if (RELU) x = fmaxf(x, 0.0f);
                h8[v] = (_Float16)x;
            }
            *(uint4*)(outH + ((size_t)img * WPO * WPO + (size_t)(oy_[s4] + 1) * WPO
                              + (ox_[s4] + 1)) * 32 + cbase) = *(const uint4*)h8;
        }
    }
}

// ---------------------------------------------------------------------------
// ECC channel descriptor: resolves (b,c) ONCE into {base, x-stride, row-stride}
// so every inner-loop access is branch-free (c==0 -> strided raw image,
// else f32 CHW feature plane).
// ---------------------------------------------------------------------------
struct ChanImg {
    const float* base;
    int xs;   // element stride in x
    int rs;   // element stride in y (row pitch)
};

__device__ __forceinline__ ChanImg chan_img(const float* __restrict__ raw,
                                            const float* __restrict__ feat,
                                            int b, int c, int W, int P, int s) {
    ChanImg ci;
    if (c == 0) { ci.base = raw + (size_t)b * 65536; ci.xs = s;  ci.rs = 256 * s; }
    else        { ci.base = feat + (size_t)(b * 32 + (c - 1)) * P; ci.xs = 1; ci.rs = W; }
    return ci;
}

__device__ __forceinline__ float cread(const ChanImg& ci, int x, int y) {
    return ci.base[y * ci.rs + x * ci.xs];
}

__device__ __forceinline__ void cgrad(const ChanImg& ci, int x, int y, int W,
                                      float* gx, float* gy) {
    float xl = cread(ci, x > 0 ? x - 1 : 0, y);
    float xr = cread(ci, x < W - 1 ? x + 1 : W - 1, y);
    float yl = cread(ci, x, y > 0 ? y - 1 : 0);
    float yr = cread(ci, x, y < W - 1 ? y + 1 : W - 1);
    *gx = (x == 0 || x == W - 1) ? (xr - xl) : 0.5f * (xr - xl);
    *gy = (y == 0 || y == W - 1) ? (yr - yl) : 0.5f * (yr - yl);
}

__device__ __forceinline__ float csample(const ChanImg& ci, int W, float xf, float yf) {
    float x0 = floorf(xf), y0 = floorf(yf);
    float wx = xf - x0, wy = yf - y0;
    int x0i = min(max((int)x0, 0), W - 1);
    int x1i = min(x0i + 1, W - 1);
    int y0i = min(max((int)y0, 0), W - 1);
    int y1i = min(y0i + 1, W - 1);
    float v00 = cread(ci, x0i, y0i);
    float v01 = cread(ci, x1i, y0i);
    float v10 = cread(ci, x0i, y1i);
    float v11 = cread(ci, x1i, y1i);
    return (1.0f - wy) * ((1.0f - wx) * v00 + wx * v01) + wy * ((1.0f - wx) * v10 + wx * v11);
}

// ---------------------------------------------------------------------------
// Template stats: sum(T), sum(T^2) per channel
// ---------------------------------------------------------------------------
__global__ void tstats_kernel(const float* __restrict__ rawT, const float* __restrict__ featT,
                              float* __restrict__ tstat, int W, int s, int shift) {
    int ch = blockIdx.x;
    int P = W * W;
    ChanImg ci = chan_img(rawT, featT, ch / 33, ch % 33, W, P, s);
    float s1 = 0.0f, s2 = 0.0f;
    for (int pix = blockIdx.y * 256 + threadIdx.x; pix < P; pix += gridDim.y * 256) {
        int x = pix & (W - 1), y = pix >> shift;
        float t = cread(ci, x, y);
        s1 += t; s2 += t * t;
    }
    __shared__ float r1[256], r2[256];
    r1[threadIdx.x] = s1; r2[threadIdx.x] = s2;
    __syncthreads();
    for (int st = 128; st > 0; st >>= 1) {
        if (threadIdx.x < st) { r1[threadIdx.x] += r1[threadIdx.x + st]; r2[threadIdx.x] += r2[threadIdx.x + st]; }
        __syncthreads();
    }
    if (threadIdx.x == 0) {
        atomicAdd(&tstat[ch * 2 + 0], r1[0]);
        atomicAdd(&tstat[ch * 2 + 1], r2[0]);
    }
}

// ---------------------------------------------------------------------------
// Hessian Hm = sum_p G_p G_p^T via v_wmma_f32_16x16x4_f32 (6 padded to 16).
// Wave stages G for 32 pixels into LDS, then 8 wmma's of K=4 pixels each.
// A and B fragments are identical (symmetric Gram matrix).
// ---------------------------------------------------------------------------
__global__ __launch_bounds__(256)
void hessian_wmma_kernel(const float* __restrict__ rawT, const float* __restrict__ featT,
                         float* __restrict__ Hm, int W, int s, int shift) {
    int ch = blockIdx.x;
    int P = W * W;
    ChanImg ci = chan_img(rawT, featT, ch / 33, ch % 33, W, P, s);
    int tid = threadIdx.x, lane = tid & 31, wave = tid >> 5;
    __shared__ float Gl[8][32][8];
    __shared__ float HmS[36];
    if (tid < 36) HmS[tid] = 0.0f;
    __syncthreads();

    v8f acc = {};
    int nChunks = P >> 8;
    int mn = lane & 15;
    int kb = (lane >> 4) * 2;

    for (int chunk = blockIdx.y; chunk < nChunks; chunk += gridDim.y) {
        int pix = (chunk << 8) + (wave << 5) + lane;
        int x = pix & (W - 1), y = pix >> shift;
        float gx, gy;
        cgrad(ci, x, y, W, &gx, &gy);
        float xs = (float)x, ys = (float)y;
        Gl[wave][lane][0] = gx * xs; Gl[wave][lane][1] = gx * ys; Gl[wave][lane][2] = gx;
        Gl[wave][lane][3] = gy * xs; Gl[wave][lane][4] = gy * ys; Gl[wave][lane][5] = gy;
        asm volatile("s_wait_dscnt 0" ::: "memory");   // wave-local LDS producer->consumer
#pragma unroll
        for (int sub = 0; sub < 8; ++sub) {
            v2f af = {};
            if (mn < 6) {
                af[0] = Gl[wave][sub * 4 + kb + 0][mn];
                af[1] = Gl[wave][sub * 4 + kb + 1][mn];
            }
            acc = __builtin_amdgcn_wmma_f32_16x16x4_f32(
                false, af, false, af, (short)0, acc, false, false);
        }
    }
#pragma unroll
    for (int v = 0; v < 8; ++v) {
        int m = v + ((lane >> 4) << 3);
        int n = lane & 15;
        if (m < 6 && n < 6) atomicAdd(&HmS[m * 6 + n], acc[v]);
    }
    __syncthreads();
    if (tid < 36) atomicAdd(&Hm[ch * 36 + tid], HmS[tid]);
}

// ---------------------------------------------------------------------------
// Warp + stats: sum(Iw), sum(Iw^2) per channel under current p
// ---------------------------------------------------------------------------
__global__ void warp_stats_kernel(const float* __restrict__ rawW, const float* __restrict__ featW,
                                  const float* __restrict__ pbuf, float* __restrict__ sumI,
                                  int W, int s, int shift) {
    int ch = blockIdx.x;
    int P = W * W;
    ChanImg ci = chan_img(rawW, featW, ch / 33, ch % 33, W, P, s);
    const float* p = pbuf + ch * 6;
    float p0 = p[0], p1 = p[1], p2 = p[2], p3 = p[3], p4 = p[4], p5 = p[5];
    float s1 = 0.0f, s2 = 0.0f;
    for (int pix = blockIdx.y * 256 + threadIdx.x; pix < P; pix += gridDim.y * 256) {
        int x = pix & (W - 1), y = pix >> shift;
        float xs = (float)x, ys = (float)y;
        float xw = p0 * xs + p1 * ys + p2;
        float yw = p3 * xs + p4 * ys + p5;
        float Iw = csample(ci, W, xw, yw);
        s1 += Iw; s2 += Iw * Iw;
    }
    __shared__ float r1[256], r2[256];
    r1[threadIdx.x] = s1; r2[threadIdx.x] = s2;
    __syncthreads();
    for (int st = 128; st > 0; st >>= 1) {
        if (threadIdx.x < st) { r1[threadIdx.x] += r1[threadIdx.x + st]; r2[threadIdx.x] += r2[threadIdx.x + st]; }
        __syncthreads();
    }
    if (threadIdx.x == 0) {
        atomicAdd(&sumI[ch * 2 + 0], r1[0]);
        atomicAdd(&sumI[ch * 2 + 1], r2[0]);
    }
}

// ---------------------------------------------------------------------------
// Residual + b = G^T err (re-samples Iw, recomputes template gradients)
// ---------------------------------------------------------------------------
__global__ void residual_b_kernel(const float* __restrict__ rawW, const float* __restrict__ featW,
                                  const float* __restrict__ rawT, const float* __restrict__ featT,
                                  const float* __restrict__ pbuf, const float* __restrict__ sumI,
                                  const float* __restrict__ tstat, float* __restrict__ bvec,
                                  int W, int s, int shift) {
    int ch = blockIdx.x;
    int P = W * W;
    ChanImg cw = chan_img(rawW, featW, ch / 33, ch % 33, W, P, s);
    ChanImg ct = chan_img(rawT, featT, ch / 33, ch % 33, W, P, s);
    float Pf = (float)P;
    float S1 = sumI[ch * 2], S2 = sumI[ch * 2 + 1];
    float muI = S1 / Pf;
    float inorm = sqrtf(fmaxf(S2 - S1 * S1 / Pf, 0.0f)) + 1e-8f;
    float T1 = tstat[ch * 2], T2 = tstat[ch * 2 + 1];
    float muT = T1 / Pf;
    float tn = sqrtf(fmaxf(T2 - T1 * T1 / Pf, 0.0f)) + 1e-8f;
    float scale = tn / inorm;
    const float* p = pbuf + ch * 6;
    float p0 = p[0], p1 = p[1], p2 = p[2], p3 = p[3], p4 = p[4], p5 = p[5];
    float bl[6] = {0, 0, 0, 0, 0, 0};
    for (int pix = blockIdx.y * 256 + threadIdx.x; pix < P; pix += gridDim.y * 256) {
        int x = pix & (W - 1), y = pix >> shift;
        float xs = (float)x, ys = (float)y;
        float xw = p0 * xs + p1 * ys + p2;
        float yw = p3 * xs + p4 * ys + p5;
        float Iw = csample(cw, W, xw, yw);
        float tv = cread(ct, x, y);
        float gx, gy;
        cgrad(ct, x, y, W, &gx, &gy);
        float err = (Iw - muI) * scale - (tv - muT);
        bl[0] += gx * xs * err; bl[1] += gx * ys * err; bl[2] += gx * err;
        bl[3] += gy * xs * err; bl[4] += gy * ys * err; bl[5] += gy * err;
    }
    __shared__ float bS[6];
    if (threadIdx.x < 6) bS[threadIdx.x] = 0.0f;
    __syncthreads();
#pragma unroll
    for (int j = 0; j < 6; ++j) atomicAdd(&bS[j], bl[j]);
    __syncthreads();
    if (threadIdx.x < 6) atomicAdd(&bvec[ch * 6 + threadIdx.x], bS[threadIdx.x]);
}

// ---------------------------------------------------------------------------
// Solve (Hm + eps I) dp = b, p -= dp. One thread per channel.
// ---------------------------------------------------------------------------
__global__ void solve_kernel(const float* __restrict__ Hm, const float* __restrict__ bvec,
                             float* __restrict__ pbuf) {
    int ch = blockIdx.x * blockDim.x + threadIdx.x;
    if (ch >= 132) return;
    float A[36], bb[6];
#pragma unroll
    for (int i = 0; i < 36; ++i) A[i] = Hm[ch * 36 + i];
#pragma unroll
    for (int d = 0; d < 6; ++d) A[d * 7] += EPS_REG;
#pragma unroll
    for (int j = 0; j < 6; ++j) bb[j] = bvec[ch * 6 + j];
    for (int k = 0; k < 6; ++k) {
        float inv = 1.0f / A[k * 6 + k];
        for (int r = k + 1; r < 6; ++r) {
            float f = A[r * 6 + k] * inv;
            for (int cc = k; cc < 6; ++cc) A[r * 6 + cc] -= f * A[k * 6 + cc];
            bb[r] -= f * bb[k];
        }
    }
    float dp[6];
    for (int k = 5; k >= 0; --k) {
        float t = bb[k];
        for (int cc = k + 1; cc < 6; ++cc) t -= A[k * 6 + cc] * dp[cc];
        dp[k] = t / A[k * 6 + k];
    }
#pragma unroll
    for (int j = 0; j < 6; ++j) pbuf[ch * 6 + j] -= dp[j];
}

// ---------------------------------------------------------------------------
// p init (identity) and final softmax aggregation MLP
// ---------------------------------------------------------------------------
__global__ void init_p_kernel(float* __restrict__ pbuf) {
    int ch = blockIdx.x * blockDim.x + threadIdx.x;
    if (ch < 132) {
        float* p = pbuf + ch * 6;
        p[0] = 1.0f; p[1] = 0.0f; p[2] = 0.0f;
        p[3] = 0.0f; p[4] = 1.0f; p[5] = 0.0f;
    }
}

__global__ void aggregate_kernel(const float* __restrict__ pbuf,
                                 const float* __restrict__ w1, const float* __restrict__ b1,
                                 const float* __restrict__ w2, const float* __restrict__ b2,
                                 float* __restrict__ out) {
    __shared__ float lg[132];
    int t = threadIdx.x;
    if (t < 132) {
        float pv[6];
#pragma unroll
        for (int j = 0; j < 6; ++j) pv[j] = pbuf[t * 6 + j];
        float logit = b2[0];
        for (int o = 0; o < 32; ++o) {
            float h = b1[o];
#pragma unroll
            for (int j = 0; j < 6; ++j) h += pv[j] * w1[j * 32 + o];
            h = fmaxf(h, 0.0f);
            logit += h * w2[o];
        }
        lg[t] = logit;
    }
    __syncthreads();
    if (t < 4) {
        float mx = -1e30f;
        for (int c = 0; c < 33; ++c) mx = fmaxf(mx, lg[t * 33 + c]);
        float den = 0.0f;
        float a6[6] = {0, 0, 0, 0, 0, 0};
        for (int c = 0; c < 33; ++c) {
            float w = expf(lg[t * 33 + c] - mx);
            den += w;
            for (int j = 0; j < 6; ++j) a6[j] += w * pbuf[(t * 33 + c) * 6 + j];
        }
        for (int j = 0; j < 6; ++j) out[t * 6 + j] = a6[j] / den;
    }
}

// ---------------------------------------------------------------------------
// Host launcher
// ---------------------------------------------------------------------------
extern "C" void kernel_launch(void* const* d_in, const int* in_sizes, int n_in,
                              void* d_out, int out_size, void* d_ws, size_t ws_size,
                              hipStream_t stream) {
    (void)in_sizes; (void)n_in; (void)out_size; (void)ws_size;
    const float* warped = (const float*)d_in[0];
    const float* templ  = (const float*)d_in[1];

    struct Lvl {
        const float *w_in, *b_in, *w_down[3], *b_down[3], *w_out, *b_out;
        int nd;
    } L[4];
    int idx = 2;
    for (int i = 0; i < 4; ++i) {
        int nd = 3 - i;
        L[i].nd = nd;
        L[i].w_in = (const float*)d_in[idx++];
        L[i].b_in = (const float*)d_in[idx++];
        for (int d = 0; d < nd; ++d) L[i].w_down[d] = (const float*)d_in[idx++];
        for (int d = 0; d < nd; ++d) L[i].b_down[d] = (const float*)d_in[idx++];
        L[i].w_out = (const float*)d_in[idx++];
        L[i].b_out = (const float*)d_in[idx++];
    }
    const float* agg_w1 = (const float*)d_in[idx++];
    const float* agg_b1 = (const float*)d_in[idx++];
    const float* agg_w2 = (const float*)d_in[idx++];
    const float* agg_b2 = (const float*)d_in[idx++];

    char* ws = (char*)d_ws;
    size_t off = 0;
    auto wsAlloc = [&](size_t bytes) -> void* {
        void* p = ws + off;
        off = (off + bytes + 255) & ~(size_t)255;
        return p;
    };
    float* featW = (float*)wsAlloc((size_t)4 * 32 * 256 * 256 * 4);
    float* featT = (float*)wsAlloc((size_t)4 * 32 * 256 * 256 * 4);
    // One f16 HWC activation buffer per resolution, each with 1-px zero halo.
    const int padW[4] = {258, 130, 66, 34};   // for W = 256,128,64,32
    _Float16* act[4];
    size_t actBytes[4];
    for (int q = 0; q < 4; ++q) {
        actBytes[q] = (size_t)4 * padW[q] * padW[q] * 32 * 2;
        act[q] = (_Float16*)wsAlloc(actBytes[q]);
    }
    _Float16* packedW = (_Float16*)wsAlloc((size_t)10 * 9216 * 2);
    float* Hm    = (float*)wsAlloc(132 * 36 * 4);
    float* sumI  = (float*)wsAlloc(132 * 2 * 4);
    float* tstat = (float*)wsAlloc(132 * 2 * 4);
    float* bvec  = (float*)wsAlloc(132 * 6 * 4);
    float* pbuf  = (float*)wsAlloc(132 * 6 * 4);

    // Zero the halos (interiors are always fully rewritten before being read).
    for (int q = 0; q < 4; ++q) hipMemsetAsync(act[q], 0, actBytes[q], stream);

    // Pack all 32->32 conv weights into WMMA fragment order (10 layers).
    const float* layerW[10];
    const float* layerB[10];
    {
        int li = 0;
        for (int lvl = 0; lvl < 4; ++lvl) {
            for (int d = 0; d < L[lvl].nd; ++d) { layerW[li] = L[lvl].w_down[d]; layerB[li] = L[lvl].b_down[d]; ++li; }
            layerW[li] = L[lvl].w_out; layerB[li] = L[lvl].b_out; ++li;
        }
        for (int l = 0; l < 10; ++l)
            pack_w32_kernel<<<36, 256, 0, stream>>>(layerW[l], packedW + (size_t)l * 9216);
    }

    init_p_kernel<<<1, 160, 0, stream>>>(pbuf);

    auto launch_down = [&](int Wi, const _Float16* in, _Float16* out,
                           const _Float16* pwv, const float* pbv) {
        int Wo = Wi / 2;
        dim3 g(Wo * Wo / 128, 4);
        switch (Wi) {
        case 256: conv32_wmma_kernel<256, 2, false, true><<<g, 128, 0, stream>>>(in, pwv, pbv, out, nullptr); break;
        case 128: conv32_wmma_kernel<128, 2, false, true><<<g, 128, 0, stream>>>(in, pwv, pbv, out, nullptr); break;
        default:  conv32_wmma_kernel< 64, 2, false, true><<<g, 128, 0, stream>>>(in, pwv, pbv, out, nullptr); break;
        }
    };
    auto launch_head = [&](int Wi, const _Float16* in, float* feat,
                           const _Float16* pwv, const float* pbv) {
        dim3 g(Wi * Wi / 128, 4);
        switch (Wi) {
        case 256: conv32_wmma_kernel<256, 1, true, false><<<g, 128, 0, stream>>>(in, pwv, pbv, nullptr, feat); break;
        case 128: conv32_wmma_kernel<128, 1, true, false><<<g, 128, 0, stream>>>(in, pwv, pbv, nullptr, feat); break;
        case  64: conv32_wmma_kernel< 64, 1, true, false><<<g, 128, 0, stream>>>(in, pwv, pbv, nullptr, feat); break;
        default:  conv32_wmma_kernel< 32, 1, true, false><<<g, 128, 0, stream>>>(in, pwv, pbv, nullptr, feat); break;
        }
    };

    int layerIdx = 0;
    for (int lvl = 0; lvl < 4; ++lvl) {
        int nd = L[lvl].nd;
        int Wl = 256 >> nd;
        int s  = 1 << nd;
        int shift = 8 - nd;
        int P = Wl * Wl;
        int levelLayerBase = layerIdx;

        for (int im = 0; im < 2; ++im) {
            const float* raw = im ? templ : warped;
            float* feat = im ? featT : featW;
            stem_kernel<<<dim3(256, 4), 256, 0, stream>>>(raw, L[lvl].w_in, L[lvl].b_in, act[0]);
            int li = layerIdx = levelLayerBase;
            int Hc = 256;
            for (int d = 0; d < nd; ++d) {
                launch_down(Hc, act[d], act[d + 1],
                            packedW + (size_t)li * 9216, layerB[li]);
                Hc /= 2; ++li;
            }
            launch_head(Hc, act[nd], feat, packedW + (size_t)li * 9216, layerB[li]);
            ++li;
            layerIdx = li;
        }

        // ---- ECC at this level ----
        int nby = (P / 256 < 16) ? (P / 256) : 16;
        hipMemsetAsync(Hm, 0, 132 * 36 * 4, stream);
        hipMemsetAsync(tstat, 0, 132 * 2 * 4, stream);
        tstats_kernel<<<dim3(132, nby), 256, 0, stream>>>(templ, featT, tstat, Wl, s, shift);
        hessian_wmma_kernel<<<dim3(132, nby), 256, 0, stream>>>(templ, featT, Hm, Wl, s, shift);
        for (int it = 0; it < 5; ++it) {
            hipMemsetAsync(sumI, 0, 132 * 2 * 4, stream);
            warp_stats_kernel<<<dim3(132, nby), 256, 0, stream>>>(warped, featW, pbuf, sumI, Wl, s, shift);
            hipMemsetAsync(bvec, 0, 132 * 6 * 4, stream);
            residual_b_kernel<<<dim3(132, nby), 256, 0, stream>>>(
                warped, featW, templ, featT, pbuf, sumI, tstat, bvec, Wl, s, shift);
            solve_kernel<<<1, 160, 0, stream>>>(Hm, bvec, pbuf);
        }
    }

    aggregate_kernel<<<1, 256, 0, stream>>>(pbuf, agg_w1, agg_b1, agg_w2, agg_b2, (float*)d_out);
}
